// Pointnet_sa_module_msg_78572131713252
// MI455X (gfx1250) — compile-verified, matrix operand order checked by reading, not run on hardware
//
#include <hip/hip_runtime.h>
#include <hip/hip_bf16.h>

// ---------------------------------------------------------------------------
// Problem constants (match reference)
// ---------------------------------------------------------------------------
#define BB      4
#define NN      8192
#define CC      64
#define NPOINT  1024
#define MTOT    (BB * NPOINT)        // 4096 centroids total
#define EPSF    0.001f
#define FPS_BLK 256
#define PTS_PER_THREAD (NN / FPS_BLK) // 32

typedef float    v8f  __attribute__((ext_vector_type(8)));
typedef _Float16 v8h  __attribute__((ext_vector_type(8)));
typedef _Float16 v16h __attribute__((ext_vector_type(16)));

static __device__ __forceinline__ v16h cat8(v8h lo, v8h hi) {
    return __builtin_shufflevector(lo, hi, 0,1,2,3,4,5,6,7,8,9,10,11,12,13,14,15);
}

// ---------------------------------------------------------------------------
// 1) Farthest point sampling: one workgroup per batch, xyz resident in LDS.
//    Writes fps_idx (int) and new_xyz (f32) directly into d_out regions.
// ---------------------------------------------------------------------------
__global__ void fps_kernel(const float* __restrict__ xyz,
                           float* __restrict__ new_xyz,
                           int*   __restrict__ fps_idx) {
    extern __shared__ float sm[];
    float* sx    = sm;                       // NN*3 floats (96 KB)
    float* red_d = sm + NN * 3;              // FPS_BLK floats
    int*   red_i = (int*)(red_d + FPS_BLK);  // FPS_BLK ints
    __shared__ int s_last;

    const int b   = blockIdx.x;
    const int tid = threadIdx.x;
    const float* xb = xyz + (size_t)b * NN * 3;

    for (int i = tid; i < NN * 3; i += FPS_BLK) sx[i] = xb[i];

    float dist[PTS_PER_THREAD];
#pragma unroll
    for (int j = 0; j < PTS_PER_THREAD; ++j) dist[j] = 1e10f;

    if (tid == 0) {
        s_last = 0;
        fps_idx[b * NPOINT] = 0;
        new_xyz[((size_t)b * NPOINT) * 3 + 0] = sx[0];
        new_xyz[((size_t)b * NPOINT) * 3 + 1] = sx[1];
        new_xyz[((size_t)b * NPOINT) * 3 + 2] = sx[2];
    }
    __syncthreads();

    for (int it = 1; it < NPOINT; ++it) {
        const int last = s_last;
        const float lx = sx[last * 3 + 0];
        const float ly = sx[last * 3 + 1];
        const float lz = sx[last * 3 + 2];

        float bd = -1.0f;
        int   bi = 0;
#pragma unroll
        for (int j = 0; j < PTS_PER_THREAD; ++j) {
            const int p = tid + j * FPS_BLK;
            const float dx = sx[p * 3 + 0] - lx;
            const float dy = sx[p * 3 + 1] - ly;
            const float dz = sx[p * 3 + 2] - lz;
            const float d  = dx * dx + dy * dy + dz * dz;
            const float m  = fminf(dist[j], d);
            dist[j] = m;
            if (m > bd) { bd = m; bi = p; }
        }
        red_d[tid] = bd;
        red_i[tid] = bi;
        __syncthreads();
        for (int s = FPS_BLK / 2; s > 0; s >>= 1) {
            if (tid < s) {
                const float od = red_d[tid + s];
                const int   oi = red_i[tid + s];
                if (od > red_d[tid] || (od == red_d[tid] && oi < red_i[tid])) {
                    red_d[tid] = od; red_i[tid] = oi;
                }
            }
            __syncthreads();
        }
        if (tid == 0) {
            const int nxt = red_i[0];
            s_last = nxt;
            fps_idx[b * NPOINT + it] = nxt;
            new_xyz[((size_t)b * NPOINT + it) * 3 + 0] = sx[nxt * 3 + 0];
            new_xyz[((size_t)b * NPOINT + it) * 3 + 1] = sx[nxt * 3 + 1];
            new_xyz[((size_t)b * NPOINT + it) * 3 + 2] = sx[nxt * 3 + 2];
        }
        __syncthreads();
    }
}

// ---------------------------------------------------------------------------
// 2) Ball query, all three radii fused; one thread per centroid.
//    Reproduces reference semantics: ascending-index first-nsample, pad with
//    first hit (or 0 if none).
// ---------------------------------------------------------------------------
__global__ void ballquery_kernel(const float* __restrict__ xyz,
                                 const float* __restrict__ new_xyz,
                                 int* __restrict__ idx0,
                                 int* __restrict__ idx1,
                                 int* __restrict__ idx2) {
    const int m = blockIdx.x * blockDim.x + threadIdx.x;
    if (m >= MTOT) return;
    const int b = m >> 10; // NPOINT == 1024
    const float* xb = xyz + (size_t)b * NN * 3;
    const float cx = new_xyz[(size_t)m * 3 + 0];
    const float cy = new_xyz[(size_t)m * 3 + 1];
    const float cz = new_xyz[(size_t)m * 3 + 2];

    int* p0 = idx0 + (size_t)m * 32;
    int* p1 = idx1 + (size_t)m * 32;
    int* p2 = idx2 + (size_t)m * 64;
    int c0 = 0, c1 = 0, c2 = 0;

    for (int n = 0; n < NN; ++n) {
        if (c0 >= 32 && c1 >= 32 && c2 >= 64) break;
        const float dx = xb[n * 3 + 0] - cx;
        const float dy = xb[n * 3 + 1] - cy;
        const float dz = xb[n * 3 + 2] - cz;
        const float d2 = dx * dx + dy * dy + dz * dz;
        if (d2 < 0.04f && c0 < 32) p0[c0++] = n;
        if (d2 < 0.16f && c1 < 32) p1[c1++] = n;
        if (d2 < 0.64f && c2 < 64) p2[c2++] = n;
    }
    const int f0 = (c0 > 0) ? p0[0] : 0;
    for (int s = c0; s < 32; ++s) p0[s] = f0;
    const int f1 = (c1 > 0) ? p1[0] : 0;
    for (int s = c1; s < 32; ++s) p1[s] = f1;
    const int f2 = (c2 > 0) ? p2[0] : 0;
    for (int s = c2; s < 64; ++s) p2[s] = f2;
}

// ---------------------------------------------------------------------------
// 3) Gather + concat (rel-xyz ++ features) into f16 activation matrix,
//    K zero-padded to 96.
// ---------------------------------------------------------------------------
__global__ void gather_kernel(const float* __restrict__ xyz,
                              const float* __restrict__ points,
                              const float* __restrict__ new_xyz,
                              const int* __restrict__ idx,
                              _Float16* __restrict__ A,
                              int ns, long long total) {
    long long e = (long long)blockIdx.x * blockDim.x + threadIdx.x;
    if (e >= total) return;
    const long long row = e / 96;
    const int c = (int)(e - row * 96);
    const int m = (int)(row / ns);
    const int b = m >> 10;
    const int n = idx[row];
    float v;
    if (c < 3) {
        v = xyz[((size_t)b * NN + n) * 3 + c] - new_xyz[(size_t)m * 3 + c];
    } else if (c < 67) {
        v = points[((size_t)b * NN + n) * CC + (c - 3)];
    } else {
        v = 0.0f;
    }
    A[e] = (_Float16)v;
}

// ---------------------------------------------------------------------------
// 4) Weight pad/convert: f32 [cout x cin] -> f16 [cout x Kpad] (zero padded)
// ---------------------------------------------------------------------------
__global__ void wpad_kernel(const float* __restrict__ w,
                            _Float16* __restrict__ wp,
                            int cout, int cin, int Kpad) {
    const int e = blockIdx.x * blockDim.x + threadIdx.x;
    if (e >= cout * Kpad) return;
    const int o = e / Kpad;
    const int k = e - o * Kpad;
    wp[e] = (k < cin) ? (_Float16)w[(size_t)o * cin + k] : (_Float16)0.0f;
}

// ---------------------------------------------------------------------------
// 5) WMMA GEMM + bias + ReLU.
//    out[row,o] = relu( sum_k A[row,k] * W[o,k] + b[o] )
//    One wave per 32x16 output tile; V_WMMA_F32_16X16X32_F16, f32 accumulate.
// ---------------------------------------------------------------------------
__global__ void gemm_relu_kernel(const _Float16* __restrict__ A,
                                 const _Float16* __restrict__ W,
                                 const float* __restrict__ bias,
                                 _Float16* __restrict__ out,
                                 int rows, int Kpad, int O) {
    const int lane = threadIdx.x & 31;
    const int half = lane >> 4;
    const int l16  = lane & 15;
    const int row0 = blockIdx.x * 32;
    const int col0 = blockIdx.y * 16;

    v8f acc0 = {};
    v8f acc1 = {};

    for (int k = 0; k < Kpad; k += 32) {
        // B fragment (weights, stored [O x Kpad] == B^T layout == same as A)
        const _Float16* wp = W + (size_t)(col0 + l16) * Kpad + k + half * 8;
        const v16h bf = cat8(*(const v8h*)wp, *(const v8h*)(wp + 16));
        // A fragments: two row tiles
        const _Float16* ap0 = A + (size_t)(row0 + l16) * Kpad + k + half * 8;
        const v16h a0 = cat8(*(const v8h*)ap0, *(const v8h*)(ap0 + 16));
        const _Float16* ap1 = A + (size_t)(row0 + 16 + l16) * Kpad + k + half * 8;
        const v16h a1 = cat8(*(const v8h*)ap1, *(const v8h*)(ap1 + 16));

        acc0 = __builtin_amdgcn_wmma_f32_16x16x32_f16(false, a0, false, bf,
                                                      (short)0, acc0, false, false);
        acc1 = __builtin_amdgcn_wmma_f32_16x16x32_f16(false, a1, false, bf,
                                                      (short)0, acc1, false, false);
    }

    const float bv = bias ? bias[col0 + l16] : 0.0f;
#pragma unroll
    for (int r = 0; r < 8; ++r) {
        const int mrow = r + half * 8;               // C/D layout: M = r + 8*(lane/16)
        float v0 = acc0[r] + bv; v0 = v0 > 0.0f ? v0 : 0.0f;
        out[(size_t)(row0 + mrow) * O + col0 + l16] = (_Float16)v0;
        float v1 = acc1[r] + bv; v1 = v1 > 0.0f ? v1 : 0.0f;
        out[(size_t)(row0 + 16 + mrow) * O + col0 + l16] = (_Float16)v1;
    }
}

// ---------------------------------------------------------------------------
// 6) BatchNorm machinery: zero, per-channel sum/sumsq, normalize.
//    Grid stride is a power of two >= O, so each thread owns one channel
//    and issues exactly two global atomics.
// ---------------------------------------------------------------------------
__global__ void zero_kernel(float* __restrict__ p, int n) {
    const int i = blockIdx.x * blockDim.x + threadIdx.x;
    if (i < n) p[i] = 0.0f;
}

__global__ void bn_stats_kernel(const _Float16* __restrict__ act,
                                float* __restrict__ sums,  // [2*O]
                                long long rows, int O) {
    const long long total  = rows * O;
    const long long stride = (long long)gridDim.x * blockDim.x; // multiple of O
    long long e = (long long)blockIdx.x * blockDim.x + threadIdx.x;
    if (e >= total) return;
    const int c = (int)(e % O); // constant across iterations (stride % O == 0)
    float s = 0.0f, q = 0.0f;
    for (; e < total; e += stride) {
        const float v = (float)act[e];
        s += v;
        q += v * v;
    }
    atomicAdd(&sums[c], s);
    atomicAdd(&sums[O + c], q);
}

__global__ void bn_apply_kernel(_Float16* __restrict__ act,
                                const float* __restrict__ sums,
                                long long rows, int O) {
    const long long total  = rows * O;
    const long long stride = (long long)gridDim.x * blockDim.x;
    const float inv_n = 1.0f / (float)rows;
    for (long long e = (long long)blockIdx.x * blockDim.x + threadIdx.x;
         e < total; e += stride) {
        const int c = (int)(e % O);
        const float mu  = sums[c] * inv_n;
        const float var = sums[O + c] * inv_n - mu * mu;
        const float v   = (float)act[e];
        act[e] = (_Float16)((v - mu) * rsqrtf(var + EPSF));
    }
}

// Final BN: f16 activations -> f32 output buffer (d_out)
__global__ void bn_apply_out_kernel(const _Float16* __restrict__ act,
                                    const float* __restrict__ sums,
                                    float* __restrict__ out,
                                    long long rows, int O) {
    const long long total  = rows * O;
    const long long stride = (long long)gridDim.x * blockDim.x;
    const float inv_n = 1.0f / (float)rows;
    for (long long e = (long long)blockIdx.x * blockDim.x + threadIdx.x;
         e < total; e += stride) {
        const int c = (int)(e % O);
        const float mu  = sums[c] * inv_n;
        const float var = sums[O + c] * inv_n - mu * mu;
        const float v   = (float)act[e];
        out[e] = (v - mu) * rsqrtf(var + EPSF);
    }
}

// ---------------------------------------------------------------------------
// 7) Max over samples -> concat buffer [4096 x 256] at column offset
// ---------------------------------------------------------------------------
__global__ void maxpool_kernel(const _Float16* __restrict__ act,
                               _Float16* __restrict__ cat,
                               int ns, int O, int coff) {
    const int e = blockIdx.x * blockDim.x + threadIdx.x;
    if (e >= MTOT * O) return;
    const int m = e / O;
    const int c = e - m * O;
    const _Float16* p = act + (size_t)m * ns * O + c;
    float best = (float)p[0];
    for (int s = 1; s < ns; ++s) {
        const float v = (float)p[(size_t)s * O];
        best = fmaxf(best, v);
    }
    cat[(size_t)m * 256 + coff + c] = (_Float16)best;
}

// ---------------------------------------------------------------------------
// Host orchestration
// ---------------------------------------------------------------------------
struct LayerCfg { int wi, bi, cin, cout, Kpad; };

extern "C" void kernel_launch(void* const* d_in, const int* in_sizes, int n_in,
                              void* d_out, int out_size, void* d_ws, size_t ws_size,
                              hipStream_t stream) {
    (void)in_sizes; (void)n_in; (void)out_size; (void)ws_size;

    const float* xyz    = (const float*)d_in[0];
    const float* points = (const float*)d_in[1];

    // d_out layout: new_xyz [4,1024,3] f32 | out [4,1024,64] f32 | fps_idx [4,1024] i32
    float* out_newxyz = (float*)d_out;
    float* out_feat   = out_newxyz + (size_t)MTOT * 3;
    int*   out_idx    = (int*)(out_feat + (size_t)MTOT * 64);

    // dict flattening: xyz, points, mlp_w[9], mlp_b[9], agg_w, agg_b
    static const LayerCfg layers[10] = {
        {2, 11, 67, 32,  96}, {3, 12, 32, 32, 32}, {4, 13, 32,  64, 32},
        {5, 14, 67, 32,  96}, {6, 15, 32, 32, 32}, {7, 16, 32,  64, 32},
        {8, 17, 67, 64,  96}, {9, 18, 64, 64, 64}, {10, 19, 64, 128, 64},
        {20, 21, 256, 64, 256}  // aggregation layer
    };

    // Workspace carving
    char*  ws  = (char*)d_ws;
    size_t off = 0;
    auto alloc = [&](size_t bytes) -> void* {
        void* p = ws + off;
        off = (off + bytes + 255) & ~(size_t)255;
        return p;
    };

    const int  nsArr[3]   = {32, 32, 64};
    const int  coffArr[3] = {0, 64, 128};
    int* idxbuf[3];
    idxbuf[0] = (int*)alloc((size_t)MTOT * 32 * sizeof(int));
    idxbuf[1] = (int*)alloc((size_t)MTOT * 32 * sizeof(int));
    idxbuf[2] = (int*)alloc((size_t)MTOT * 64 * sizeof(int));

    _Float16* wpadp[10];
    for (int l = 0; l < 10; ++l)
        wpadp[l] = (_Float16*)alloc((size_t)layers[l].cout * layers[l].Kpad * sizeof(_Float16));

    float*    stats = (float*)alloc(2 * 128 * sizeof(float));
    _Float16* catb  = (_Float16*)alloc((size_t)MTOT * 256 * sizeof(_Float16));
    _Float16* bufA  = (_Float16*)alloc((size_t)262144 * 128 * sizeof(_Float16));
    _Float16* bufB  = (_Float16*)alloc((size_t)262144 * 128 * sizeof(_Float16));

    // ---- FPS (writes new_xyz + fps_idx into d_out) ----
    const size_t fps_shmem = (size_t)(NN * 3 + FPS_BLK) * sizeof(float) + FPS_BLK * sizeof(int);
    fps_kernel<<<BB, FPS_BLK, fps_shmem, stream>>>(xyz, out_newxyz, out_idx);

    // ---- weight pad/convert (independent of FPS) ----
    for (int l = 0; l < 10; ++l) {
        const LayerCfg& L = layers[l];
        const int n = L.cout * L.Kpad;
        wpad_kernel<<<(n + 255) / 256, 256, 0, stream>>>(
            (const float*)d_in[L.wi], wpadp[l], L.cout, L.cin, L.Kpad);
    }

    // ---- ball query (all radii) ----
    ballquery_kernel<<<(MTOT + 255) / 256, 256, 0, stream>>>(
        xyz, out_newxyz, idxbuf[0], idxbuf[1], idxbuf[2]);

    // ---- per-scale MLP stacks ----
    int li = 0;
    for (int s = 0; s < 3; ++s) {
        const int ns = nsArr[s];
        const long long rows = (long long)MTOT * ns;

        const long long gtotal = rows * 96;
        gather_kernel<<<(unsigned)((gtotal + 255) / 256), 256, 0, stream>>>(
            xyz, points, out_newxyz, idxbuf[s], bufA, ns, gtotal);

        _Float16* cur = bufA;
        _Float16* nxt = bufB;
        int coutLast = 0;
        for (int l = 0; l < 3; ++l, ++li) {
            const LayerCfg& L = layers[li];
            dim3 grid((unsigned)(rows / 32), (unsigned)(L.cout / 16));
            gemm_relu_kernel<<<grid, 32, 0, stream>>>(
                cur, wpadp[li], (const float*)d_in[L.bi], nxt,
                (int)rows, L.Kpad, L.cout);

            zero_kernel<<<1, 256, 0, stream>>>(stats, 2 * L.cout);
            bn_stats_kernel<<<512, 256, 0, stream>>>(nxt, stats, rows, L.cout);
            bn_apply_kernel<<<2048, 256, 0, stream>>>(nxt, stats, rows, L.cout);

            _Float16* t = cur; cur = nxt; nxt = t;
            coutLast = L.cout;
        }
        maxpool_kernel<<<(MTOT * coutLast + 255) / 256, 256, 0, stream>>>(
            cur, catb, ns, coutLast, coffArr[s]);
    }

    // ---- aggregation: [4096 x 256] -> [4096 x 64], ReLU, BN -> f32 out ----
    {
        const LayerCfg& L = layers[9];
        dim3 grid(MTOT / 32, L.cout / 16);
        gemm_relu_kernel<<<grid, 32, 0, stream>>>(
            catb, wpadp[9], (const float*)d_in[L.bi], bufA,
            MTOT, L.Kpad, L.cout);

        zero_kernel<<<1, 256, 0, stream>>>(stats, 2 * L.cout);
        bn_stats_kernel<<<512, 256, 0, stream>>>(bufA, stats, (long long)MTOT, L.cout);
        bn_apply_out_kernel<<<256, 256, 0, stream>>>(bufA, stats, out_feat,
                                                     (long long)MTOT, L.cout);
    }
}
